// GConv_55482387530255
// MI455X (gfx1250) — matrix operand order repaired and checked by reading, not compiled, not run on hardware
//
#include <hip/hip_runtime.h>
#include <stdint.h>

// ---------------- problem constants ----------------
#define BB    8
#define SS    96
#define BS    768          // B*S
#define NN    200          // sensors / nodes
#define DD    128          // feature dim
#define NP    224          // node dim padded to multiple of 32 (K of graph GEMM)
#define NPR   256          // padded adjacency rows
#define RTOT  153600       // BS*NN rows
#define TH_   0.01f
#define EPS_  1e-5f

typedef __attribute__((ext_vector_type(16))) __bf16 v16bf;
typedef __attribute__((ext_vector_type(2)))  __bf16 v2bf;
typedef __attribute__((ext_vector_type(8)))  float  v8f;
typedef __attribute__((ext_vector_type(2)))  float  v2f;
typedef __attribute__((ext_vector_type(4)))  unsigned int v4u;

union BFrag { v16bf v; v4u q[2]; unsigned int d[8]; };

// hardware packed float2 -> bf16x2 (v_cvt_pk_bf16_f32 on gfx1250)
__device__ inline unsigned int pack_bf16(float lo, float hi) {
  v2f a; a.x = lo; a.y = hi;
  v2bf b = __builtin_convertvector(a, v2bf);
  return __builtin_bit_cast(unsigned int, b);
}
__device__ inline unsigned short f2bf(float f) {
  __bf16 b = (__bf16)f;
  return __builtin_bit_cast(unsigned short, b);
}

// A-operand fragment from row-major [M][K] bf16, tile at (m0,k0). ld = K stride.
// VGPR j: k = (lane<16?0:16)+2j ; m = lane%16 (contiguous k order, matches B)
__device__ inline v16bf load_frag_A(const unsigned short* M, int ld, int m0, int k0, int lane) {
  const unsigned short* p = M + (size_t)(m0 + (lane & 15)) * ld + k0 + ((lane & 16) ? 16 : 0);
  BFrag f;
  f.q[0] = *reinterpret_cast<const v4u*>(p);
  f.q[1] = *reinterpret_cast<const v4u*>(p + 8);
  return f.v;
}

// B-operand fragment from [N][K] bf16 (K contiguous), tile at (k0,n0).
// VGPR j: k = (lane<16?0:16)+2j ; n = lane%16   (contiguous k order, matches A)
__device__ inline v16bf load_frag_B(const unsigned short* Mt, int ld, int n0, int k0, int lane) {
  const unsigned short* p = Mt + (size_t)(n0 + (lane & 15)) * ld + k0 + ((lane & 16) ? 16 : 0);
  BFrag f;
  f.q[0] = *reinterpret_cast<const v4u*>(p);
  f.q[1] = *reinterpret_cast<const v4u*>(p + 8);
  return f.v;
}

__device__ inline v8f wmma_bf16(v16bf a, v16bf b, v8f c) {
  return __builtin_amdgcn_wmma_f32_16x16x32_bf16(false, a, false, b, (short)0, c, false, false);
}

// ---------------- prep kernels (tiny) ----------------

__global__ void k_zero_stats(float* stats) {
  int t = threadIdx.x;
  if (t < 512) stats[t] = 0.0f;
}

// adaptive map: mean over (batch, heads) of attn_s[0]
__global__ void k_adaptive_mean(const float* __restrict__ attn, float* __restrict__ M1) {
  int t = blockIdx.x * blockDim.x + threadIdx.x;
  if (t >= NN * NN) return;
  float s = 0.f;
  #pragma unroll 4
  for (int k = 0; k < 64; ++k) s += attn[(size_t)k * NN * NN + t];
  M1[t] = s * (1.0f / 64.0f);
}

// symmetrize + abs-threshold each row, store Aproc; rsqrt of (rowsum+1) per row
__global__ void k_process_rows(const float* __restrict__ supports,
                               const float* __restrict__ M1,
                               float* __restrict__ Aproc, float* __restrict__ drow) {
  int m = blockIdx.x, imap = blockIdx.y, tid = threadIdx.x;
  const float* raw = (imap == 0) ? supports : M1;
  float val = 0.f;
  if (tid < NN) {
    float rv = (m >= tid) ? raw[m * NN + tid] : raw[tid * NN + m];
    float av = fabsf(rv);
    val = (av > TH_) ? av : 0.f;
    Aproc[((size_t)imap * NN + m) * NN + tid] = val;
  }
  __shared__ float red[256];
  red[tid] = val;
  __syncthreads();
  for (int o = 128; o > 0; o >>= 1) {
    if (tid < o) red[tid] += red[tid + o];
    __syncthreads();
  }
  if (tid == 0) drow[imap * NN + m] = rsqrtf(red[0] + 1.0f);  // +1 self loop
}

// An_p[i][m][n] = d[m]*d[n]*(Aproc + I), bf16, zero-padded to [NPR][NP]
__global__ void k_normalize_adj(const float* __restrict__ Aproc,
                                const float* __restrict__ drow,
                                unsigned short* __restrict__ Anp) {
  int t = blockIdx.x * blockDim.x + threadIdx.x;
  if (t >= 2 * NPR * NP) return;
  int imap = t / (NPR * NP);
  int rem  = t % (NPR * NP);
  int m = rem / NP, n = rem % NP;
  float v = 0.f;
  if (m < NN && n < NN) {
    float a = Aproc[((size_t)imap * NN + m) * NN + n] + ((m == n) ? 1.0f : 0.0f);
    v = drow[imap * NN + m] * drow[imap * NN + n] * a;
  }
  Anp[t] = f2bf(v);
}

// WeffT[i][g][k] = sum_l W_conv[i,l,k,g]; WoutT[d][c] = W_out[c][d]; beff = sum_l b_conv
__global__ void k_prep_weights(const float* __restrict__ Wc, const float* __restrict__ bc,
                               const float* __restrict__ Wout,
                               unsigned short* __restrict__ WeffT,
                               unsigned short* __restrict__ WoutT,
                               float* __restrict__ beff) {
  int t = blockIdx.x * blockDim.x + threadIdx.x;
  if (t < 2 * DD * DD) {
    int imap = t >> 14, rem = t & 16383;
    int g = rem >> 7, k = rem & 127;
    float w = Wc[((size_t)(imap * 2 + 0) * DD + k) * DD + g] +
              Wc[((size_t)(imap * 2 + 1) * DD + k) * DD + g];
    WeffT[((size_t)imap * DD + g) * DD + k] = f2bf(w);
  } else if (t < 4 * DD * DD) {
    int t2 = t - 2 * DD * DD;          // t2 = d*256 + c
    WoutT[t2] = f2bf(Wout[(size_t)(t2 & 255) * DD + (t2 >> 8)]);
  }
  if (t < 2 * DD) {
    int imap = t >> 7, g = t & 127;
    beff[t] = bc[(size_t)(imap * 2 + 0) * DD + g] + bc[(size_t)(imap * 2 + 1) * DD + g];
  }
}

// B-operand fragment built directly from fp32 x rows (hardware packed cvt to bf16).
// x is [B][N][S][D]; for fixed (b,s) the "node" rows have D contiguous.
// node columns >= NN are clamped (their products are discarded downstream).
__device__ inline v16bf load_frag_B_x(const float* __restrict__ x, int b, int s,
                                      int n0, int k0, int lane) {
  int node = n0 + (lane & 15);
  if (node >= NN) node = NN - 1;
  const float* p = x + ((size_t)(b * NN + node) * SS + s) * DD + k0 + ((lane & 16) ? 16 : 0);
  const float4 u0 = *reinterpret_cast<const float4*>(p);
  const float4 u1 = *reinterpret_cast<const float4*>(p + 4);
  const float4 u2 = *reinterpret_cast<const float4*>(p + 8);
  const float4 u3 = *reinterpret_cast<const float4*>(p + 12);
  BFrag f;
  f.d[0] = pack_bf16(u0.x, u0.y); f.d[1] = pack_bf16(u0.z, u0.w);
  f.d[2] = pack_bf16(u1.x, u1.y); f.d[3] = pack_bf16(u1.z, u1.w);
  f.d[4] = pack_bf16(u2.x, u2.y); f.d[5] = pack_bf16(u2.z, u2.w);
  f.d[6] = pack_bf16(u3.x, u3.y); f.d[7] = pack_bf16(u3.z, u3.w);
  return f.v;
}

// ---------------- GEMM 1: Zt[bs][g][node] = (Weff_i^T) x (x[b,:,s,:]^T) ----------------
// M = g (128), N = node (store 224 cols; cols>=200 stored as 0), K = 128
__global__ __launch_bounds__(256) void k_feat_gemm(const float* __restrict__ x,
                                                   const unsigned short* __restrict__ WeffT,
                                                   unsigned short* __restrict__ Zt, int imap) {
  int bs = blockIdx.y;
  int b = bs / SS, s = bs % SS;
  int lane = threadIdx.x & 31, w = threadIdx.x >> 5;
  int m0 = (w & 3) * 32;                       // g
  int n0 = blockIdx.x * 64 + (w >> 2) * 32;    // node
  const unsigned short* A = WeffT + (size_t)imap * DD * DD;
  v8f acc[2][2] = {};
  for (int kk = 0; kk < DD; kk += 32) {
    v16bf a0 = load_frag_A(A, DD, m0,      kk, lane);
    v16bf a1 = load_frag_A(A, DD, m0 + 16, kk, lane);
    v16bf b0 = load_frag_B_x(x, b, s, n0,      kk, lane);
    v16bf b1 = load_frag_B_x(x, b, s, n0 + 16, kk, lane);
    acc[0][0] = wmma_bf16(a0, b0, acc[0][0]);
    acc[0][1] = wmma_bf16(a0, b1, acc[0][1]);
    acc[1][0] = wmma_bf16(a1, b0, acc[1][0]);
    acc[1][1] = wmma_bf16(a1, b1, acc[1][1]);
  }
  unsigned short* Z = Zt + (size_t)bs * DD * NP;
  int half = lane >> 4, cl = lane & 15;
  #pragma unroll
  for (int mi = 0; mi < 2; ++mi)
    #pragma unroll
    for (int ni = 0; ni < 2; ++ni) {
      int node = n0 + ni * 16 + cl;
      if (node >= NP) continue;
      #pragma unroll
      for (int j = 0; j < 8; ++j) {
        int g = m0 + mi * 16 + j + half * 8;
        float v = (node < NN) ? acc[mi][ni][j] : 0.0f;   // exact zero K-padding
        Z[(size_t)g * NP + node] = f2bf(v);
      }
    }
}

// ---------------- GEMM 2: h_i[bs][node][g] = An_i x Zt[bs] + beff ----------------
// M = node (200 of 256 padded), N = g (128), K = node-padded (224)
// BN sum/sumsq per channel are accumulated here (LDS reduce -> global atomics).
__global__ __launch_bounds__(256) void k_graph_gemm(const unsigned short* __restrict__ Anp,
                                                    const unsigned short* __restrict__ Zt,
                                                    const float* __restrict__ beff,
                                                    float* __restrict__ hout,
                                                    float* __restrict__ stats, int imap) {
  int bs = blockIdx.z;
  int lane = threadIdx.x & 31, w = threadIdx.x >> 5;
  int m0 = blockIdx.y * 128 + (w & 3) * 32;     // node row
  int n0 = blockIdx.x * 64 + (w >> 2) * 32;     // g
  const unsigned short* A  = Anp + (size_t)imap * NPR * NP;
  const unsigned short* Bt = Zt + (size_t)bs * DD * NP;
  v8f acc[2][2] = {};
  for (int kk = 0; kk < NP; kk += 32) {
    if (kk + 32 < NP)
      __builtin_prefetch(Bt + (size_t)(n0 + (lane & 15)) * NP + kk + 32, 0, 3);
    v16bf a0 = load_frag_A(A, NP, m0,      kk, lane);
    v16bf a1 = load_frag_A(A, NP, m0 + 16, kk, lane);
    v16bf b0 = load_frag_B(Bt, NP, n0,      kk, lane);
    v16bf b1 = load_frag_B(Bt, NP, n0 + 16, kk, lane);
    acc[0][0] = wmma_bf16(a0, b0, acc[0][0]);
    acc[0][1] = wmma_bf16(a0, b1, acc[0][1]);
    acc[1][0] = wmma_bf16(a1, b0, acc[1][0]);
    acc[1][1] = wmma_bf16(a1, b1, acc[1][1]);
  }
  __shared__ float sSum[64], sSq[64];
  if (threadIdx.x < 64) { sSum[threadIdx.x] = 0.f; sSq[threadIdx.x] = 0.f; }
  __syncthreads();
  int half = lane >> 4, cl = lane & 15;
  #pragma unroll
  for (int ni = 0; ni < 2; ++ni) {
    int g = n0 + ni * 16 + cl;
    float bias = beff[imap * DD + g];
    float ps = 0.f, pq = 0.f;
    #pragma unroll
    for (int mi = 0; mi < 2; ++mi)
      #pragma unroll
      for (int j = 0; j < 8; ++j) {
        int m = m0 + mi * 16 + j + half * 8;
        if (m >= NN) continue;
        float v = acc[mi][ni][j] + bias;
        hout[((size_t)bs * NN + m) * DD + g] = v;
        ps += v; pq += v * v;
      }
    int lg = (w >> 2) * 32 + ni * 16 + cl;     // channel local to block (0..63)
    atomicAdd(&sSum[lg], ps);
    atomicAdd(&sSq[lg], pq);
  }
  __syncthreads();
  if (threadIdx.x < 64) {
    int g = blockIdx.x * 64 + threadIdx.x;
    atomicAdd(&stats[imap * 256 + g], sSum[threadIdx.x]);
    atomicAdd(&stats[imap * 256 + 128 + g], sSq[threadIdx.x]);
  }
}

// scale/shift: ss[i][0][g]=gamma*rsqrt(var+eps), ss[i][1][g]=beta-mean*scale
__global__ void k_bn_final(const float* __restrict__ stats, const float* __restrict__ gamma,
                           const float* __restrict__ beta, float* __restrict__ ss) {
  int t = threadIdx.x;
  if (t >= 256) return;
  int imap = t >> 7, g = t & 127;
  const float inv = 1.0f / (float)RTOT;
  float mean = stats[imap * 256 + g] * inv;
  float var  = stats[imap * 256 + 128 + g] * inv - mean * mean;
  var = fmaxf(var, 0.f);
  float sc = gamma[imap * DD + g] * rsqrtf(var + EPS_);
  ss[imap * 256 + g]       = sc;
  ss[imap * 256 + 128 + g] = beta[imap * DD + g] - mean * sc;
}

// build bf16 A-fragment of cat = relu(bn(h)) on the fly (ssL lives in LDS)
__device__ inline v16bf make_cat_fragA(const float* __restrict__ h,
                                       const float* ssL,
                                       size_t r0, int k0, int lane) {
  BFrag f;
  int half = lane >> 4;
  size_t row = r0 + (lane & 15);
  int base = k0 + (half ? 16 : 0);
  #pragma unroll
  for (int chunk = 0; chunk < 2; ++chunk) {
    int cb = base + chunk * 8;             // 8 contiguous channels, never crosses map bdry
    int imap = cb >> 7, g = cb & 127;
    const float* hp  = h + ((size_t)imap * RTOT + row) * DD + g;
    const float* scp = ssL + imap * 256 + g;
    const float* shp = scp + 128;
    #pragma unroll
    for (int p = 0; p < 4; ++p) {
      float v0 = fmaxf(hp[2 * p]     * scp[2 * p]     + shp[2 * p],     0.f);
      float v1 = fmaxf(hp[2 * p + 1] * scp[2 * p + 1] + shp[2 * p + 1], 0.f);
      f.d[chunk * 4 + p] = pack_bf16(v0, v1);
    }
  }
  return f.v;
}

// ---------------- final GEMM: out = relu(cat x W_out + b), with inverse reshape ----------------
// M = r (153600), N = d (128), K = 256
__global__ __launch_bounds__(256) void k_out_gemm(const float* __restrict__ h,
                                                  const float* __restrict__ ss,
                                                  const unsigned short* __restrict__ WoutT,
                                                  const float* __restrict__ b_out,
                                                  float* __restrict__ out) {
  __shared__ float ssL[512];
  for (int t = threadIdx.x; t < 512; t += 256) ssL[t] = ss[t];
  __syncthreads();
  int lane = threadIdx.x & 31, w = threadIdx.x >> 5;
  size_t r0 = (size_t)blockIdx.y * 128 + (size_t)(w & 3) * 32;
  int d0 = blockIdx.x * 64 + (w >> 2) * 32;
  v8f acc[2][2] = {};
  for (int kk = 0; kk < 256; kk += 32) {
    v16bf a0 = make_cat_fragA(h, ssL, r0,      kk, lane);
    v16bf a1 = make_cat_fragA(h, ssL, r0 + 16, kk, lane);
    v16bf b0 = load_frag_B(WoutT, 256, d0,      kk, lane);
    v16bf b1 = load_frag_B(WoutT, 256, d0 + 16, kk, lane);
    acc[0][0] = wmma_bf16(a0, b0, acc[0][0]);
    acc[0][1] = wmma_bf16(a0, b1, acc[0][1]);
    acc[1][0] = wmma_bf16(a1, b0, acc[1][0]);
    acc[1][1] = wmma_bf16(a1, b1, acc[1][1]);
  }
  int half = lane >> 4, cl = lane & 15;
  #pragma unroll
  for (int mi = 0; mi < 2; ++mi)
    #pragma unroll
    for (int ni = 0; ni < 2; ++ni) {
      int d = d0 + ni * 16 + cl;
      float bias = b_out[d];
      #pragma unroll
      for (int j = 0; j < 8; ++j) {
        size_t r = r0 + mi * 16 + j + half * 8;          // always < RTOT (1200*128 exact)
        float y = fmaxf(acc[mi][ni][j] + bias, 0.f);
        int bsI = (int)(r / NN), n = (int)(r % NN);
        int b = bsI / SS, s = bsI % SS;
        out[(((size_t)(b * NN + n)) * SS + s) * DD + d] = y;
      }
    }
}

// ---------------- host-side orchestration ----------------
extern "C" void kernel_launch(void* const* d_in, const int* in_sizes, int n_in,
                              void* d_out, int out_size, void* d_ws, size_t ws_size,
                              hipStream_t stream) {
  (void)in_sizes; (void)n_in; (void)out_size; (void)ws_size;
  const float* x        = (const float*)d_in[0];
  const float* attn_s   = (const float*)d_in[1];
  const float* supports = (const float*)d_in[2];
  const float* W_conv   = (const float*)d_in[3];
  const float* b_conv   = (const float*)d_in[4];
  const float* bn_gamma = (const float*)d_in[5];
  const float* bn_beta  = (const float*)d_in[6];
  const float* W_out    = (const float*)d_in[7];
  const float* b_out    = (const float*)d_in[8];
  float* out = (float*)d_out;

  uintptr_t p = (uintptr_t)d_ws;
  auto take = [&](size_t bytes) { uintptr_t r = (p + 255) & ~(uintptr_t)255; p = r + bytes; return r; };
  unsigned short* Zt    = (unsigned short*)take((size_t)BS * DD * NP * 2);
  float*          h     = (float*)take((size_t)2 * RTOT * DD * 4);
  unsigned short* Anp   = (unsigned short*)take((size_t)2 * NPR * NP * 2);
  float*          Aproc = (float*)take((size_t)2 * NN * NN * 4);
  float*          drow  = (float*)take((size_t)2 * NN * 4);
  float*          M1    = (float*)take((size_t)NN * NN * 4);
  unsigned short* WeffT = (unsigned short*)take((size_t)2 * DD * DD * 2);
  unsigned short* WoutT = (unsigned short*)take((size_t)DD * 256 * 2);
  float*          beff  = (float*)take((size_t)2 * DD * 4);
  float*          stats = (float*)take((size_t)512 * 4);
  float*          ss    = (float*)take((size_t)512 * 4);

  // prep
  k_zero_stats<<<1, 512, 0, stream>>>(stats);
  k_adaptive_mean<<<(NN * NN + 255) / 256, 256, 0, stream>>>(attn_s, M1);
  k_process_rows<<<dim3(NN, 2), 256, 0, stream>>>(supports, M1, Aproc, drow);
  k_normalize_adj<<<(2 * NPR * NP + 255) / 256, 256, 0, stream>>>(Aproc, drow, Anp);
  k_prep_weights<<<(4 * DD * DD + 255) / 256, 256, 0, stream>>>(W_conv, b_conv, W_out,
                                                                WeffT, WoutT, beff);
  // main pipeline: Zt buffer is reused per map (stream-ordered)
  for (int imap = 0; imap < 2; ++imap) {
    k_feat_gemm<<<dim3(4, BS), 256, 0, stream>>>(x, WeffT, Zt, imap);
    k_graph_gemm<<<dim3(2, 2, BS), 256, 0, stream>>>(Anp, Zt, beff,
                                                     h + (size_t)imap * RTOT * DD,
                                                     stats, imap);
  }
  k_bn_final<<<1, 256, 0, stream>>>(stats, bn_gamma, bn_beta, ss);
  k_out_gemm<<<dim3(2, 1200), 256, 0, stream>>>(h, ss, WoutT, b_out, out);
}
